// RNN_3221225472312
// MI455X (gfx1250) — compile-verified, matrix operand order checked by reading, not run on hardware
//
#include <hip/hip_runtime.h>
#include <math.h>

#define X_DIM 128
#define H_DIM 512
#define C_DIM 256
#define Y_DIM 128
#define BATCH 64
#define SEQ   512
#define BLOCK 512   // 16 waves: 4 per SIMD for latency hiding

typedef __attribute__((ext_vector_type(16))) __bf16        v16bf;
typedef __attribute__((ext_vector_type(8)))  float         v8f;
typedef __attribute__((ext_vector_type(4)))  float         f32x4;
typedef __attribute__((ext_vector_type(4)))  unsigned int  u32x4;

// Explicit global-address-space pointer types: keep streamed operand loads on
// the global_load path (FLAT loads would also tick DScnt and serialize with
// the LDS operand stream). Ext-vector types (not HIP_vector_type classes) so
// copies from AS(1) lvalues are plain loads in both host and device passes.
typedef __attribute__((address_space(1))) const u32x4  gu32x4;
typedef __attribute__((address_space(1))) const f32x4  gf32x4;

union ABReg { v16bf v; u32x4 q[2]; __bf16 e[16]; };

#if __has_builtin(__builtin_amdgcn_tanhf)
__device__ inline float fast_tanh(float v) { return __builtin_amdgcn_tanhf(v); }
#else
__device__ inline float fast_tanh(float v) { return tanhf(v); }
#endif

__device__ inline v8f vzero8() {
    v8f z;
#pragma unroll
    for (int i = 0; i < 8; ++i) z[i] = 0.0f;
    return z;
}

__device__ inline v8f wmma_bf16(v16bf a, v16bf b, v8f c) {
    return __builtin_amdgcn_wmma_f32_16x16x32_bf16(
        /*neg_a=*/false, a, /*neg_b=*/false, b,
        /*c_mod=*/(short)0, c, /*reuse_a=*/false, /*reuse_b=*/false);
}

// Pre-swizzled bf16 operand block in LDS (512 elems = 1KB): lane's 32B chunk.
__device__ inline v16bf load_sw(const __bf16* blk, int lane) {
    ABReg u;
    const u32x4* p = (const u32x4*)(blk + (lane << 4));
    u.q[0] = p[0];
    u.q[1] = p[1];
    return u.v;
}

// Pre-swizzled bf16 operand block in GLOBAL memory (forced addrspace(1)).
__device__ inline v16bf load_sw_g(const __bf16* blk, int lane) {
    ABReg u;
    gu32x4* p = (gu32x4*)(blk + (lane << 4));
    u.q[0] = p[0];
    u.q[1] = p[1];
    return u.v;
}

// 16-bit WMMA operand from row-major bf16 LDS row pointer.
// ISA layout: lane {idx=lane&15, kg=(lane>>4)*8}; elems 0..7 = K kg..kg+7,
// elems 8..15 = K kg+16..kg+23 (relative to colbase).
__device__ inline v16bf load_lds16(const __bf16* row, int colbase_kg) {
    ABReg u;
    u.q[0] = *(const u32x4*)(row + colbase_kg);
    u.q[1] = *(const u32x4*)(row + colbase_kg + 16);
    return u.v;
}

// Same layout, converting from f32 GLOBAL memory on the fly (p -> K base).
__device__ inline v16bf cvt_f32_16(const float* p) {
    gf32x4* q = (gf32x4*)p;
    f32x4 a = q[0], b = q[1], c = q[4], d = q[5];
    ABReg u;
    u.e[0]  = (__bf16)a[0]; u.e[1]  = (__bf16)a[1]; u.e[2]  = (__bf16)a[2]; u.e[3]  = (__bf16)a[3];
    u.e[4]  = (__bf16)b[0]; u.e[5]  = (__bf16)b[1]; u.e[6]  = (__bf16)b[2]; u.e[7]  = (__bf16)b[3];
    u.e[8]  = (__bf16)c[0]; u.e[9]  = (__bf16)c[1]; u.e[10] = (__bf16)c[2]; u.e[11] = (__bf16)c[3];
    u.e[12] = (__bf16)d[0]; u.e[13] = (__bf16)d[1]; u.e[14] = (__bf16)d[2]; u.e[15] = (__bf16)d[3];
    return u.v;
}

// Pre-swizzle a row-major f32 weight (K x N) into the per-lane WMMA B layout.
__global__ void swizzle_weight(const float* __restrict__ W, __bf16* __restrict__ dst,
                               int K, int N) {
    int idx = blockIdx.x * blockDim.x + threadIdx.x;
    if (idx >= K * N) return;
    int nkb    = K >> 5;
    int e      = idx & 15;
    int L      = (idx >> 4) & 31;
    int blk    = idx >> 9;
    int kb     = blk % nkb;
    int n_tile = blk / nkb;
    int n  = n_tile * 16 + (L & 15);
    int kg = (L >> 4) * 8;
    int k  = kb * 32 + kg + (e < 8 ? e : e + 8);
    dst[idx] = (__bf16)W[(size_t)k * N + n];
}

// Pre-swizzle Ad (SEQ x C x C, f32) into bf16 B-operand blocks for c @ A_t^T:
// B_t[k][n] = A_t[n][k]. Per t: 16 n-tiles x 8 kb x 512-elem blocks (128 KB).
__global__ void swizzle_ad(const float* __restrict__ Ad, __bf16* __restrict__ dst) {
    size_t idx = (size_t)blockIdx.x * blockDim.x + threadIdx.x;  // 512*65536 total
    int within = (int)(idx & 65535);
    int t      = (int)(idx >> 16);
    int e   = within & 15;
    int L   = (within >> 4) & 31;
    int blk = within >> 9;          // 0..127
    int kb  = blk & 7, nt = blk >> 3;
    int n   = nt * 16 + (L & 15);
    int kg  = (L >> 4) * 8;
    int k   = kb * 32 + kg + (e < 8 ? e : e + 8);
    dst[idx] = (__bf16)Ad[((size_t)t << 16) + (size_t)n * C_DIM + k];
}

__global__ void __launch_bounds__(BLOCK, 1)
hippo_rnn_kernel(const float* __restrict__ x,  const float* __restrict__ bh,
                 const float* __restrict__ by, const float* __restrict__ fw,
                 const float* __restrict__ fb, const float* __restrict__ Ad,
                 const float* __restrict__ bd,
                 const __bf16* __restrict__ wxh_sw, const __bf16* __restrict__ whh_sw,
                 const __bf16* __restrict__ wyh_sw, const __bf16* __restrict__ ad_sw,
                 float* __restrict__ out) {
    // ---- 304 KB of the WGP's 320 KB LDS ----
    __shared__ __attribute__((aligned(16))) __bf16 h_buf[2][16 * H_DIM];   //  32 KB
    __shared__ __attribute__((aligned(16))) __bf16 c_buf[2][16 * C_DIM];   //  16 KB
    __shared__ __attribute__((aligned(16))) __bf16 lds_wyh[8 * 16 * 512];  // 128 KB
    __shared__ __attribute__((aligned(16))) __bf16 lds_wxh[32 * 4 * 512];  // 128 KB (x-slice)
    __shared__ float f_lds[16];

    const int tid  = threadIdx.x;
    const int w    = tid >> 5;           // wave id 0..15
    const int lane = tid & 31;
    const int am   = lane & 15;          // A-operand M row / B-operand N col
    const int akg  = (lane >> 4) * 8;    // K sub-group base within 32-block
    const int hi   = lane >> 4;          // C/D layout: row = v + 8*hi
    const int n0   = lane & 15;          // C/D layout: col
    const int b0   = blockIdx.x * 16;    // batch slice base

    // Zero state buffers (buffer 0 holds h_0 = 0, c_0 = 0).
    for (int i = tid; i < 16 * H_DIM; i += BLOCK) h_buf[0][i] = (__bf16)0.0f;
    for (int i = tid; i < 16 * C_DIM; i += BLOCK) c_buf[0][i] = (__bf16)0.0f;

    // Stage Wyh (all 8*16 blocks) into LDS, 16B vector copies.
    for (int i = tid; i < (8 * 16 * 512) / 8; i += BLOCK)
        ((u32x4*)lds_wyh)[i] = ((gu32x4*)wyh_sw)[i];
    // Stage the x-slice of Wxh (kb 0..3 of each of 32 n-tiles) into LDS.
    for (int i = tid; i < (32 * 4 * 512) / 8; i += BLOCK) {
        int blk = i >> 6;                 // 64 u32x4 per 512-elem block
        int off = i & 63;
        int nt  = blk >> 2, kb = blk & 3;
        ((u32x4*)lds_wxh)[i] =
            ((gu32x4*)(wxh_sw + ((size_t)(nt * 12 + kb) << 9)))[off];
    }
    __syncthreads();

    const float fb0 = fb[0];
    const float* xrow = x + (size_t)(b0 + am) * SEQ * X_DIM;

    // Opaque zero offset, redefined every timestep: blocks LICM from hoisting
    // loop-invariant B-operand blocks into VGPRs (which would spill), while
    // keeping the base pointers as kernel args (provably global -> no FLAT).
    size_t lzero = 0;

    for (int t = 0; t < SEQ; ++t) {
        asm volatile("" : "+s"(lzero));
        const __bf16* wxh_p = wxh_sw + lzero;
        const __bf16* whh_p = whh_sw + lzero;
        const int r = t & 1, wb = r ^ 1;
        if (tid < 16) f_lds[tid] = 0.0f;

        // ---- Phase A: h_new = tanh([x_t, c] @ Wxh + h @ Whh + bh) ----
        // 32 N-tiles over 16 waves: wave w owns tiles {w, w+16}; one shared
        // A operand feeds 2 independent accumulators per wave, and 4 waves
        // per SIMD cover the LDS/L2 load latency.
        v8f acc[2];
#pragma unroll
        for (int j = 0; j < 2; ++j) acc[j] = vzero8();

        const float* xp = xrow + (size_t)t * X_DIM;
#pragma unroll
        for (int kb = 0; kb < 4; ++kb) {   // K 0..127: x part, weights in LDS
            v16bf a = cvt_f32_16(xp + kb * 32 + akg);
#pragma unroll
            for (int j = 0; j < 2; ++j)
                acc[j] = wmma_bf16(a,
                    load_sw(&lds_wxh[(size_t)((w + 16 * j) * 4 + kb) << 9], lane), acc[j]);
        }
#pragma unroll
        for (int kb = 0; kb < 8; ++kb) {   // K 128..383: c part, weights stream from L2
            v16bf a = load_lds16(&c_buf[r][am * C_DIM], kb * 32 + akg);
#pragma unroll
            for (int j = 0; j < 2; ++j)
                acc[j] = wmma_bf16(a,
                    load_sw_g(wxh_p + ((size_t)((w + 16 * j) * 12 + kb + 4) << 9), lane), acc[j]);
        }
#pragma unroll
        for (int kb = 0; kb < 16; ++kb) {  // K = 512: recurrent h, Whh from L2
            v16bf a = load_lds16(&h_buf[r][am * H_DIM], kb * 32 + akg);
#pragma unroll
            for (int j = 0; j < 2; ++j)
                acc[j] = wmma_bf16(a,
                    load_sw_g(whh_p + ((size_t)((w + 16 * j) * 16 + kb) << 9), lane), acc[j]);
        }
#pragma unroll
        for (int j = 0; j < 2; ++j) {
            int ng = (w + 16 * j) * 16 + n0;
            float bhv = bh[ng];
#pragma unroll
            for (int v = 0; v < 8; ++v) {
                int m = v + 8 * hi;
                h_buf[wb][m * H_DIM + ng] = (__bf16)fast_tanh(acc[j][v] + bhv);
            }
        }
        __syncthreads();

        // ---- Phase B: f[m] = sum_k h[m][k] * fw[k]  (LDS atomic reduction) ----
        {
            int row = tid & 15, ch = tid >> 4;   // 32 chunks of 16 columns
            const __bf16* hp = &h_buf[wb][row * H_DIM + ch * 16];
            const float*  fp = fw + ch * 16;
            float s = 0.0f;
#pragma unroll 8
            for (int i = 0; i < 16; ++i) s += (float)hp[i] * fp[i];
            atomicAdd(&f_lds[row], s);
        }
        __syncthreads();

        // ---- Phases C & D run concurrently on disjoint wave sets ----
        if (w < 8) {
            // Phase C (waves 0-7): y = h @ Wyh + by, 1 N-tile per wave, LDS-fed.
            v8f yacc = vzero8();
#pragma unroll
            for (int kb = 0; kb < 16; ++kb) {
                v16bf a = load_lds16(&h_buf[wb][am * H_DIM], kb * 32 + akg);
                yacc = wmma_bf16(a, load_sw(&lds_wyh[(size_t)(w * 16 + kb) << 9], lane), yacc);
            }
            int ng = w * 16 + n0;
            float byv = by[ng];
#pragma unroll
            for (int v = 0; v < 8; ++v) {
                int m = v + 8 * hi;
                out[((size_t)(b0 + m) * SEQ + t) * Y_DIM + ng] = yacc[v] + byv;
            }
        } else {
            // Phase D (waves 8-15): c_new = c @ A_t^T + f * b_t^T, L2-fed.
            const int wd = w - 8;
            v8f cacc[2];
#pragma unroll
            for (int j = 0; j < 2; ++j) cacc[j] = vzero8();

            if (ad_sw != nullptr) {
                // Pre-swizzled bf16 A_t: direct B-operand loads, no converts.
                const __bf16* At_sw = ad_sw + ((size_t)t << 16);
#pragma unroll
                for (int kb = 0; kb < 8; ++kb) {
                    v16bf a = load_lds16(&c_buf[r][am * C_DIM], kb * 32 + akg);
#pragma unroll
                    for (int j = 0; j < 2; ++j) {
                        int nt = wd + 8 * j;
                        v16bf b = load_sw_g(At_sw + ((size_t)(nt * 8 + kb) << 9), lane);
                        cacc[j] = wmma_bf16(a, b, cacc[j]);
                    }
                }
            } else {
                // Fallback: convert f32 A_t rows on the fly.
                const float* At = Ad + (size_t)t * (C_DIM * C_DIM);
#pragma unroll
                for (int kb = 0; kb < 8; ++kb) {
                    v16bf a = load_lds16(&c_buf[r][am * C_DIM], kb * 32 + akg);
#pragma unroll
                    for (int j = 0; j < 2; ++j) {
                        int ng = (wd + 8 * j) * 16 + n0;
                        v16bf b = cvt_f32_16(At + (size_t)ng * C_DIM + kb * 32 + akg);
                        cacc[j] = wmma_bf16(a, b, cacc[j]);
                    }
                }
            }

            const float* bdt = bd + (size_t)t * C_DIM;
#pragma unroll
            for (int j = 0; j < 2; ++j) {
                int ng = (wd + 8 * j) * 16 + n0;
                float bdv = bdt[ng];
#pragma unroll
                for (int v = 0; v < 8; ++v) {
                    int m = v + 8 * hi;
                    float fv = f_lds[m] + fb0;
                    c_buf[wb][m * C_DIM + ng] = (__bf16)(cacc[j][v] + fv * bdv);
                }
            }
            if (t + 1 < SEQ) {  // hint A_{t+1} into cache while pipes drain
                int pid = tid - 256;  // 0..255 within phase-D waves
                if (ad_sw != nullptr)
                    __builtin_prefetch((const char*)(ad_sw + ((size_t)(t + 1) << 16)) + pid * 512, 0, 1);
                else
                    __builtin_prefetch(Ad + (size_t)(t + 1) * (C_DIM * C_DIM) + pid * 256, 0, 1);
            }
        }
        __syncthreads();
    }
}

extern "C" void kernel_launch(void* const* d_in, const int* in_sizes, int n_in,
                              void* d_out, int out_size, void* d_ws, size_t ws_size,
                              hipStream_t stream) {
    (void)in_sizes; (void)n_in; (void)out_size;
    const float* x   = (const float*)d_in[0];
    const float* Wxh = (const float*)d_in[1];
    const float* Whh = (const float*)d_in[2];
    const float* bh  = (const float*)d_in[3];
    const float* Wyh = (const float*)d_in[4];
    const float* by  = (const float*)d_in[5];
    const float* fw  = (const float*)d_in[6];
    const float* fb  = (const float*)d_in[7];
    const float* Ad  = (const float*)d_in[8];
    const float* bd  = (const float*)d_in[9];

    // Workspace layout: 1 MB pre-swizzled bf16 weights [+ 64 MB bf16 Ad].
    const size_t WXH_E = (size_t)(X_DIM + C_DIM) * H_DIM;  // 196608
    const size_t WHH_E = (size_t)H_DIM * H_DIM;            // 262144
    const size_t WYH_E = (size_t)H_DIM * Y_DIM;            // 65536
    const size_t W_E   = WXH_E + WHH_E + WYH_E;            // 524288 elems (1 MB)
    const size_t AD_E  = (size_t)SEQ * C_DIM * C_DIM;      // 33554432 elems (64 MB)

    __bf16* wxh_sw = (__bf16*)d_ws;
    __bf16* whh_sw = wxh_sw + WXH_E;
    __bf16* wyh_sw = whh_sw + WHH_E;
    const bool have_ad = ws_size >= (W_E + AD_E) * sizeof(__bf16);
    __bf16* ad_sw = have_ad ? (wxh_sw + W_E) : nullptr;

    swizzle_weight<<<(WXH_E + 255) / 256, 256, 0, stream>>>(Wxh, wxh_sw, X_DIM + C_DIM, H_DIM);
    swizzle_weight<<<(WHH_E + 255) / 256, 256, 0, stream>>>(Whh, whh_sw, H_DIM, H_DIM);
    swizzle_weight<<<(WYH_E + 255) / 256, 256, 0, stream>>>(Wyh, wyh_sw, H_DIM, Y_DIM);
    if (have_ad)
        swizzle_ad<<<(unsigned)(AD_E / 256), 256, 0, stream>>>(Ad, ad_sw);

    hippo_rnn_kernel<<<BATCH / 16, BLOCK, 0, stream>>>(
        x, bh, by, fw, fb, Ad, bd, wxh_sw, whh_sw, wyh_sw, ad_sw, (float*)d_out);
}
